// HeteroGAT_59854664237708
// MI455X (gfx1250) — matrix operand order repaired and checked by reading, not compiled, not run on hardware
//
#include <hip/hip_runtime.h>

// ---------------------------------------------------------------------------
// HeteroGAT for MI455X (gfx1250).
// GEMMs: bf16 WMMA, pre-converted bf16 activations + fragment-packed weights.
//        One wave = 32 rows x 128 cols (2 A-fragments x 8 B-tiles = 16 WMMA
//        per k-step); every B fragment load feeds two WMMAs.
// Edge phases: bandwidth kernels with L2-resident float atomics.
// ---------------------------------------------------------------------------

#define DEVFN __device__ __forceinline__

typedef __attribute__((ext_vector_type(16))) __bf16 bf16x16;
typedef __attribute__((ext_vector_type(8)))  __bf16 bf16x8;
typedef __attribute__((ext_vector_type(8)))  float  floatx8;

// Problem sizes (fixed by the reference)
constexpr size_t NSpot = 50000;
constexpr size_t NUser = 100000;
constexpr size_t NEdge = 1000000;
constexpr size_t HC    = 128;

// ---------------- workspace layout (float offsets) -------------------------
constexpr size_t OFF_XS_POOL = 0;                               // [NS,512]
constexpr size_t OFF_A       = OFF_XS_POOL + NSpot * 512;       // scratch region A
// region A alias #1: pooling GEMM output [NS*5, 512]
constexpr size_t OFF_HPOOL   = OFF_A;
constexpr size_t SZ_REGION_A = NSpot * 5 * 512;                 // 128M floats
// region A alias #2 (after pooling): per-layer GAT buffers
constexpr size_t OFF_H_VISIT_SRC = OFF_A;                        // [NU,128]
constexpr size_t OFF_H_VISIT_DST = OFF_H_VISIT_SRC + NUser * HC; // [NS,128]
constexpr size_t OFF_H_NEAR_SRC  = OFF_H_VISIT_DST + NSpot * HC;
constexpr size_t OFF_H_NEAR_DST  = OFF_H_NEAR_SRC + NSpot * HC;
constexpr size_t OFF_H_REV_SRC   = OFF_H_NEAR_DST + NSpot * HC;
constexpr size_t OFF_H_REV_DST   = OFF_H_REV_SRC + NSpot * HC;   // [NU,128]
constexpr size_t OFF_AV_SRC      = OFF_H_REV_DST + NUser * HC;   // [NU]
constexpr size_t OFF_AV_DST      = OFF_AV_SRC + NUser;           // [NS]
constexpr size_t OFF_AN_SRC      = OFF_AV_DST + NSpot;
constexpr size_t OFF_AN_DST      = OFF_AN_SRC + NSpot;
constexpr size_t OFF_AR_SRC      = OFF_AN_DST + NSpot;
constexpr size_t OFF_AR_DST      = OFF_AR_SRC + NSpot;           // [NU]
constexpr size_t OFF_SEGMAX      = OFF_AR_DST + NUser;           // [NU] (shared)
constexpr size_t OFF_SEGSUM      = OFF_SEGMAX + NUser;
constexpr size_t OFF_EDGE_E      = OFF_SEGSUM + NUser;           // [E]
constexpr size_t OFF_EDGE_EX     = OFF_EDGE_E + NEdge;           // [E]
constexpr size_t OFF_OUT_VISIT   = OFF_EDGE_EX + NEdge;          // [NS,128]
constexpr size_t OFF_OUT_NEAR    = OFF_OUT_VISIT + NSpot * HC;
constexpr size_t OFF_OUT_REV     = OFF_OUT_NEAR + NSpot * HC;    // [NU,128]
constexpr size_t END_A           = OFF_OUT_REV + NUser * HC;
static_assert(END_A <= OFF_A + SZ_REGION_A, "region A overflow");
// persistent region B
constexpr size_t OFF_B       = OFF_A + SZ_REGION_A;
constexpr size_t OFF_COLL_S0 = OFF_B;
constexpr size_t OFF_COLL_S1 = OFF_COLL_S0 + NSpot * HC;
constexpr size_t OFF_COLL_U0 = OFF_COLL_S1 + NSpot * HC;
constexpr size_t OFF_COLL_U1 = OFF_COLL_U0 + NUser * HC;
constexpr size_t OFF_XS_CUR  = OFF_COLL_U1 + NUser * HC;
constexpr size_t OFF_XU_CUR  = OFF_XS_CUR + NSpot * HC;
// region C: bf16 operands (offsets below are in __bf16 elements from BF base)
constexpr size_t OFF_BF_BASE = OFF_XU_CUR + NUser * HC;          // float offset
constexpr size_t BOFF_XSPOT  = 0;                                // NS*2560 bf16
constexpr size_t BOFF_XUSER  = BOFF_XSPOT + NSpot * 2560;        // NU*256
constexpr size_t BOFF_XSFEAT = BOFF_XUSER + NUser * 256;         // NS*512
constexpr size_t BOFF_XSCUR  = BOFF_XSFEAT + NSpot * 512;        // NS*128
constexpr size_t BOFF_XUCUR  = BOFF_XSCUR + NSpot * HC;          // NU*128
constexpr size_t BOFF_BPATT  = BOFF_XUCUR + NUser * HC;          // 512*512
constexpr size_t BOFF_BPW    = BOFF_BPATT + 512 * 512;           // 12 packed weights

// ---------------------------------------------------------------------------
DEVFN __bf16 f2bf(float f) {
  unsigned u = __float_as_uint(f);
  u += 0x7FFFu + ((u >> 16) & 1u);       // round-to-nearest-even
  unsigned short h = (unsigned short)(u >> 16);
  __bf16 r;
  __builtin_memcpy(&r, &h, sizeof(r));
  return r;
}

// fp32 -> bf16 bulk conversion, 8 elements/thread (n % 8 == 0).
__global__ void cvt_bf16(const float* __restrict__ x, __bf16* __restrict__ y, size_t n)
{
  size_t i = ((size_t)blockIdx.x * blockDim.x + threadIdx.x) * 8;
  if (i >= n) return;
  bf16x8 v;
#pragma unroll
  for (int e = 0; e < 8; ++e) v[e] = f2bf(x[i + e]);
  *(bf16x8*)(y + i) = v;
}

// Pack row-major fp32 B[K,N] into WMMA B-fragment order:
// Bp[((tile*(K/32)+kk)*32 + lane)*16 + e] = B[kk*32 + (lane>>4)*16 + e][tile*16 + (lane&15)]
__global__ void pack_b(const float* __restrict__ B, __bf16* __restrict__ Bp,
                       int K, int N)
{
  const int ksteps = K >> 5;
  const int total = (N / 16) * ksteps * 32;
  const int idx = blockIdx.x * blockDim.x + threadIdx.x;
  if (idx >= total) return;
  const int lane = idx & 31;
  const int kk   = (idx >> 5) % ksteps;
  const int tile = idx / (32 * ksteps);
  const int col  = tile * 16 + (lane & 15);
  const int kb   = kk * 32 + (lane >> 4) * 16;
  bf16x16 v;
#pragma unroll
  for (int e = 0; e < 16; ++e) v[e] = f2bf(B[(size_t)(kb + e) * N + col]);
  *(bf16x16*)(Bp + (size_t)idx * 16) = v;
}

// Pack att_W [4,512,128] as virtual B[512,512] (col n = h*128+c) fragments.
__global__ void pack_attW(const float* __restrict__ W, __bf16* __restrict__ Bp)
{
  const int ksteps = 16;                  // K=512
  const int idx = blockIdx.x * blockDim.x + threadIdx.x;
  if (idx >= 32 * ksteps * 32) return;    // 32 N-tiles
  const int lane = idx & 31;
  const int kk   = (idx >> 5) % ksteps;
  const int tile = idx / (32 * ksteps);
  const int n    = tile * 16 + (lane & 15);
  const int h = n >> 7, c = n & 127;
  const int kb = kk * 32 + (lane >> 4) * 16;
  bf16x16 v;
#pragma unroll
  for (int e = 0; e < 16; ++e)
    v[e] = f2bf(W[((size_t)h * 512 + (size_t)(kb + e)) * 128 + c]);
  *(bf16x16*)(Bp + (size_t)idx * 16) = v;
}

// C[M,N] = A[M,K] @ B[K,N]; A bf16 row-major, B pre-packed fragments, C fp32.
// One wave computes MT*16 rows x NT*16 cols; each B fragment load feeds MT
// WMMAs. K%32==0, M%16==0 (tail row-fragment guarded when M%(MT*16)!=0).
template<int NT, int MT>
__global__ __launch_bounds__(32)
void gemm_bf16p(const __bf16* __restrict__ A, const __bf16* __restrict__ Bp,
                float* __restrict__ C, int M, int K, int lda, int ldc)
{
  const int lane = threadIdx.x;
  const int half = lane >> 4;
  const int l15  = lane & 15;
  const int tile0 = blockIdx.x * NT;
  const size_t m0 = (size_t)blockIdx.y * (16 * MT);
  const int ksteps = K >> 5;

  const __bf16* arow[MT];
  bool mvalid[MT];
#pragma unroll
  for (int m = 0; m < MT; ++m) {
    const size_t r = m0 + (size_t)m * 16;
    mvalid[m] = (r < (size_t)M);
    arow[m] = A + (mvalid[m] ? (r + l15) : (size_t)l15) * (size_t)lda;
  }

  floatx8 acc[MT][NT] = {};
  for (int kk = 0; kk < ksteps; ++kk) {
    bf16x16 a[MT];
#pragma unroll
    for (int m = 0; m < MT; ++m) {
      const __bf16* ap = arow[m] + kk * 32 + half * 16;
      bf16x8 alo = *(const bf16x8*)(ap);        // K = kk*32 + 16*half + 0..7
      bf16x8 ahi = *(const bf16x8*)(ap + 8);    // K = kk*32 + 16*half + 8..15
      a[m] = __builtin_shufflevector(alo, ahi,
          0, 1, 2, 3, 4, 5, 6, 7, 8, 9, 10, 11, 12, 13, 14, 15);
    }
#pragma unroll
    for (int t = 0; t < NT; ++t) {
      const bf16x16 b = *(const bf16x16*)(
          Bp + ((size_t)(tile0 + t) * ksteps + kk) * (32 * 16) + (size_t)lane * 16);
#pragma unroll
      for (int m = 0; m < MT; ++m)
        acc[m][t] = __builtin_amdgcn_wmma_f32_16x16x32_bf16(false, a[m], false, b,
                                                            (short)0, acc[m][t],
                                                            false, false);
    }
  }
#pragma unroll
  for (int m = 0; m < MT; ++m) {
    if (!mvalid[m]) continue;
#pragma unroll
    for (int t = 0; t < NT; ++t) {
      const size_t col = (size_t)(tile0 + t) * 16 + l15;
#pragma unroll
      for (int i = 0; i < 8; ++i) {
        const size_t r = m0 + (size_t)m * 16 + (size_t)(half * 8 + i);
        C[r * (size_t)ldc + col] = acc[m][t][i];
      }
    }
  }
}

// Attention pooling tail: hpool [NS*5, 512] (col = h*128+c) -> xs [NS, 512].
__global__ __launch_bounds__(128)
void att_pool_finish(const float* __restrict__ hpool, const float* __restrict__ q,
                     float* __restrict__ xs)
{
  __shared__ float red[128];
  __shared__ float score[4][5];
  const size_t n = blockIdx.x;
  const int c = threadIdx.x;
  for (int h = 0; h < 4; ++h) {
    for (int s = 0; s < 5; ++s) {
      red[c] = hpool[(n * 5 + s) * 512 + h * 128 + c] * q[h * 128 + c];
      __syncthreads();
      for (int off = 64; off > 0; off >>= 1) {
        if (c < off) red[c] += red[c + off];
        __syncthreads();
      }
      if (c == 0) {
        float e = red[0];
        score[h][s] = (e > 0.f) ? e : 0.2f * e;   // leaky_relu(0.2)
      }
      __syncthreads();
    }
  }
  for (int h = 0; h < 4; ++h) {
    float m = score[h][0];
#pragma unroll
    for (int s = 1; s < 5; ++s) m = fmaxf(m, score[h][s]);
    float w[5], den = 0.f;
#pragma unroll
    for (int s = 0; s < 5; ++s) { w[s] = __expf(score[h][s] - m); den += w[s]; }
    float accum = 0.f;
#pragma unroll
    for (int s = 0; s < 5; ++s)
      accum += (w[s] / den) * hpool[(n * 5 + s) * 512 + h * 128 + c];
    xs[n * 512 + h * 128 + c] = accum;
  }
}

// a[i] = dot(H[i, 0:128], w). One wave per row, float4 per lane.
__global__ __launch_bounds__(256)
void rowdot128(const float* __restrict__ H, const float* __restrict__ w,
               float* __restrict__ a, int nrows)
{
  const int row  = blockIdx.x * 8 + (threadIdx.x >> 5);
  const int lane = threadIdx.x & 31;
  if (row >= nrows) return;
  const float4 hv = ((const float4*)(H + (size_t)row * HC))[lane];
  const float4 wv = ((const float4*)w)[lane];
  float s = hv.x * wv.x + hv.y * wv.y + hv.z * wv.z + hv.w * wv.w;
#pragma unroll
  for (int off = 16; off > 0; off >>= 1) s += __shfl_xor(s, off, 32);
  if (lane == 0) a[row] = s;
}

__global__ void fill_f32(float* __restrict__ p, float v, size_t n) {
  size_t i = (size_t)blockIdx.x * blockDim.x + threadIdx.x;
  if (i < n) p[i] = v;
}

// e = leaky_relu(a_src[src] + a_dst[dst]); segmax[dst] = max(segmax[dst], e)
__global__ void edge_logits(const float* __restrict__ a_src,
                            const float* __restrict__ a_dst,
                            const int* __restrict__ src, const int* __restrict__ dst,
                            float* __restrict__ e, float* __restrict__ segmax, int ne)
{
  int i = blockIdx.x * blockDim.x + threadIdx.x;
  if (i >= ne) return;
  float v = a_src[src[i]] + a_dst[dst[i]];
  v = (v > 0.f) ? v : 0.2f * v;
  e[i] = v;
  float* addr = &segmax[dst[i]];
  if (v >= 0.f) atomicMax((int*)addr, __float_as_int(v));
  else          atomicMin((unsigned int*)addr, __float_as_uint(v));
}

__global__ void edge_exp(const float* __restrict__ e, const int* __restrict__ dst,
                         const float* __restrict__ segmax, float* __restrict__ ex,
                         float* __restrict__ segsum, int ne)
{
  int i = blockIdx.x * blockDim.x + threadIdx.x;
  if (i >= ne) return;
  float v = __expf(e[i] - segmax[dst[i]]);
  ex[i] = v;
  atomicAdd(&segsum[dst[i]], v);
}

// out[dst] += (ex/segsum[dst]) * h_src[src].  One wave per edge, float4/lane.
__global__ __launch_bounds__(256)
void edge_scatter(const float* __restrict__ ex, const float* __restrict__ segsum,
                  const float* __restrict__ h_src,
                  const int* __restrict__ src, const int* __restrict__ dst,
                  float* __restrict__ out, int ne)
{
  const int e    = blockIdx.x * 8 + (threadIdx.x >> 5);
  const int lane = threadIdx.x & 31;
  if (e >= ne) return;
  const int s = src[e], d = dst[e];
  const float alpha = ex[e] / (segsum[d] + 1e-16f);
  const float4 v = ((const float4*)(h_src + (size_t)s * HC))[lane];
  float* o = out + (size_t)d * HC + lane * 4;
  atomicAdd(o + 0, alpha * v.x);
  atomicAdd(o + 1, alpha * v.y);
  atomicAdd(o + 2, alpha * v.z);
  atomicAdd(o + 3, alpha * v.w);
}

// HeteroConv mean over 2 spot edge types (+bias each) and ReLU copy.
__global__ void combine_spot(const float* __restrict__ ov, const float* __restrict__ bv,
                             const float* __restrict__ on, const float* __restrict__ bn,
                             float* __restrict__ coll, float* __restrict__ xrelu, size_t n)
{
  size_t i = (size_t)blockIdx.x * blockDim.x + threadIdx.x;
  if (i >= n) return;
  const int c = (int)(i & (HC - 1));
  float v = 0.5f * ((ov[i] + bv[c]) + (on[i] + bn[c]));
  coll[i] = v;
  xrelu[i] = fmaxf(v, 0.f);
}

__global__ void combine_user(const float* __restrict__ o, const float* __restrict__ b,
                             float* __restrict__ coll, float* __restrict__ xrelu, size_t n)
{
  size_t i = (size_t)blockIdx.x * blockDim.x + threadIdx.x;
  if (i >= n) return;
  const int c = (int)(i & (HC - 1));
  float v = o[i] + b[c];
  coll[i] = v;
  xrelu[i] = fmaxf(v, 0.f);
}

__global__ void mean2(const float* __restrict__ a, const float* __restrict__ b,
                      float* __restrict__ o, size_t n)
{
  size_t i = (size_t)blockIdx.x * blockDim.x + threadIdx.x;
  if (i < n) o[i] = 0.5f * (a[i] + b[i]);
}

// out[row] = dot(X[row,0:128], W) + b[0].  One wave per row.
__global__ __launch_bounds__(256)
void final_linear(const float* __restrict__ X, const float* __restrict__ W,
                  const float* __restrict__ b, float* __restrict__ out, int nrows)
{
  const int row  = blockIdx.x * 8 + (threadIdx.x >> 5);
  const int lane = threadIdx.x & 31;
  if (row >= nrows) return;
  const float4 xv = ((const float4*)(X + (size_t)row * HC))[lane];
  const float4 wv = ((const float4*)W)[lane];
  float s = xv.x * wv.x + xv.y * wv.y + xv.z * wv.z + xv.w * wv.w;
#pragma unroll
  for (int off = 16; off > 0; off >>= 1) s += __shfl_xor(s, off, 32);
  if (lane == 0) out[row] = s + b[0];
}

// ---------------------------------------------------------------------------
extern "C" void kernel_launch(void* const* d_in, const int* in_sizes, int n_in,
                              void* d_out, int out_size, void* d_ws, size_t ws_size,
                              hipStream_t stream)
{
  (void)in_sizes; (void)n_in; (void)out_size; (void)ws_size;
  float*  ws = (float*)d_ws;
  __bf16* bf = (__bf16*)(ws + OFF_BF_BASE);

  // Inputs in setup_inputs() dict order (nested dicts in insertion order).
  const float* x_spot = (const float*)d_in[0];   // [NS,2560] == [NS*5,512]
  const float* x_user = (const float*)d_in[1];   // [NU,256]
  const float* att_W  = (const float*)d_in[2];   // [4,512,128]
  const float* att_q  = (const float*)d_in[3];   // [4,128]
  auto GP = [&](int l, int et, int k) -> const float* {
    return (const float*)d_in[4 + (l * 3 + et) * 5 + k];   // et: 0=visit 1=rev 2=near
  };
  const float* spot_W = (const float*)d_in[34];
  const float* spot_b = (const float*)d_in[35];
  const float* user_W = (const float*)d_in[36];
  const float* user_b = (const float*)d_in[37];
  const int* e_src[3] = { (const int*)d_in[38], (const int*)d_in[40], (const int*)d_in[42] };
  const int* e_dst[3] = { (const int*)d_in[39], (const int*)d_in[41], (const int*)d_in[43] };

  const int NSi = (int)NSpot, NUi = (int)NUser, NEi = (int)NEdge;
  const float NEG_INF = -__builtin_inff();

  // ---- 0) Convert activations to bf16; pack all weight matrices.
  cvt_bf16<<<(unsigned)((NSpot * 2560 / 8 + 255) / 256), 256, 0, stream>>>(
      x_spot, bf + BOFF_XSPOT, NSpot * 2560);
  cvt_bf16<<<(unsigned)((NUser * 256 / 8 + 255) / 256), 256, 0, stream>>>(
      x_user, bf + BOFF_XUSER, NUser * 256);
  pack_attW<<<(32 * 16 * 32 + 255) / 256, 256, 0, stream>>>(att_W, bf + BOFF_BPATT);
  // 12 GAT weights: w = (l*3+et)*2 + role  (role 0 = W_src, 1 = W_dst)
  const int KW[12] = {256, 512, 512, 256, 512, 512, 128, 128, 128, 128, 128, 128};
  __bf16* bpw[12];
  {
    size_t off = BOFF_BPW;
    for (int w = 0; w < 12; ++w) { bpw[w] = bf + off; off += (size_t)KW[w] * 128; }
  }
  for (int w = 0; w < 12; ++w) {
    const int total = 8 * KW[w];   // (128/16)*(K/32)*32
    pack_b<<<(total + 255) / 256, 256, 0, stream>>>(
        (const float*)d_in[4 + (w / 2) * 5 + (w % 2)], bpw[w], KW[w], 128);
  }

  // ---- 1) Attention pooling: single fused GEMM [NS*5,512]x[512,512].
  gemm_bf16p<8, 2><<<dim3(4, (unsigned)((NSpot * 5 + 31) / 32)), 32, 0, stream>>>(
      bf + BOFF_XSPOT, bf + BOFF_BPATT, ws + OFF_HPOOL,
      /*M=*/(int)(NSpot * 5), /*K=*/512, /*lda=*/512, /*ldc=*/512);
  att_pool_finish<<<(unsigned)NSpot, 128, 0, stream>>>(
      ws + OFF_HPOOL, att_q, ws + OFF_XS_POOL);
  cvt_bf16<<<(unsigned)((NSpot * 512 / 8 + 255) / 256), 256, 0, stream>>>(
      ws + OFF_XS_POOL, bf + BOFF_XSFEAT, NSpot * 512);

  // ---- 2) Two GAT layers over 3 edge types.
  const __bf16* xs_bf = bf + BOFF_XSFEAT; int Ks = 512;
  const __bf16* xu_bf = bf + BOFF_XUSER;  int Ku = 256;
  for (int l = 0; l < 2; ++l) {
    auto gemm = [&](const __bf16* A, int M, int K, const __bf16* Bp, float* C) {
      gemm_bf16p<8, 2><<<dim3(1, (unsigned)((M + 31) / 32)), 32, 0, stream>>>(
          A, Bp, C, M, K, K, (int)HC);
    };
    gemm(xu_bf, NUi, Ku, bpw[(l * 3 + 0) * 2 + 0], ws + OFF_H_VISIT_SRC);
    gemm(xs_bf, NSi, Ks, bpw[(l * 3 + 0) * 2 + 1], ws + OFF_H_VISIT_DST);
    gemm(xs_bf, NSi, Ks, bpw[(l * 3 + 1) * 2 + 0], ws + OFF_H_REV_SRC);
    gemm(xu_bf, NUi, Ku, bpw[(l * 3 + 1) * 2 + 1], ws + OFF_H_REV_DST);
    gemm(xs_bf, NSi, Ks, bpw[(l * 3 + 2) * 2 + 0], ws + OFF_H_NEAR_SRC);
    gemm(xs_bf, NSi, Ks, bpw[(l * 3 + 2) * 2 + 1], ws + OFF_H_NEAR_DST);

    auto rdot = [&](size_t hoff, const float* w, size_t aoff, int n) {
      rowdot128<<<(n + 7) / 8, 256, 0, stream>>>(ws + hoff, w, ws + aoff, n);
    };
    rdot(OFF_H_VISIT_SRC, GP(l, 0, 2), OFF_AV_SRC, NUi);
    rdot(OFF_H_VISIT_DST, GP(l, 0, 3), OFF_AV_DST, NSi);
    rdot(OFF_H_REV_SRC,   GP(l, 1, 2), OFF_AR_SRC, NSi);
    rdot(OFF_H_REV_DST,   GP(l, 1, 3), OFF_AR_DST, NUi);
    rdot(OFF_H_NEAR_SRC,  GP(l, 2, 2), OFF_AN_SRC, NSi);
    rdot(OFF_H_NEAR_DST,  GP(l, 2, 3), OFF_AN_DST, NSi);

    // Edge passes (sequential; segmax/segsum/edge scratch shared).
    struct Job { int et; size_t as, ad, hs, out; int ndst; };
    const Job jobs[3] = {
      {0, OFF_AV_SRC, OFF_AV_DST, OFF_H_VISIT_SRC, OFF_OUT_VISIT, NSi},
      {2, OFF_AN_SRC, OFF_AN_DST, OFF_H_NEAR_SRC,  OFF_OUT_NEAR,  NSi},
      {1, OFF_AR_SRC, OFF_AR_DST, OFF_H_REV_SRC,   OFF_OUT_REV,   NUi},
    };
    for (int j = 0; j < 3; ++j) {
      const Job& J = jobs[j];
      const size_t nacc = (size_t)J.ndst * HC;
      fill_f32<<<(unsigned)((J.ndst + 255) / 256), 256, 0, stream>>>(
          ws + OFF_SEGMAX, NEG_INF, (size_t)J.ndst);
      fill_f32<<<(unsigned)((J.ndst + 255) / 256), 256, 0, stream>>>(
          ws + OFF_SEGSUM, 0.f, (size_t)J.ndst);
      fill_f32<<<(unsigned)((nacc + 255) / 256), 256, 0, stream>>>(
          ws + J.out, 0.f, nacc);
      edge_logits<<<(NEi + 255) / 256, 256, 0, stream>>>(
          ws + J.as, ws + J.ad, e_src[J.et], e_dst[J.et],
          ws + OFF_EDGE_E, ws + OFF_SEGMAX, NEi);
      edge_exp<<<(NEi + 255) / 256, 256, 0, stream>>>(
          ws + OFF_EDGE_E, e_dst[J.et], ws + OFF_SEGMAX,
          ws + OFF_EDGE_EX, ws + OFF_SEGSUM, NEi);
      edge_scatter<<<(NEi + 7) / 8, 256, 0, stream>>>(
          ws + OFF_EDGE_EX, ws + OFF_SEGSUM, ws + J.hs,
          e_src[J.et], e_dst[J.et], ws + J.out, NEi);
    }

    // HeteroConv aggregation + ReLU; stash pre-ReLU per-layer outputs.
    const size_t coll_s = (l == 0) ? OFF_COLL_S0 : OFF_COLL_S1;
    const size_t coll_u = (l == 0) ? OFF_COLL_U0 : OFF_COLL_U1;
    const size_t ns_el = NSpot * HC, nu_el = NUser * HC;
    combine_spot<<<(unsigned)((ns_el + 255) / 256), 256, 0, stream>>>(
        ws + OFF_OUT_VISIT, GP(l, 0, 4), ws + OFF_OUT_NEAR, GP(l, 2, 4),
        ws + coll_s, ws + OFF_XS_CUR, ns_el);
    combine_user<<<(unsigned)((nu_el + 255) / 256), 256, 0, stream>>>(
        ws + OFF_OUT_REV, GP(l, 1, 4), ws + coll_u, ws + OFF_XU_CUR, nu_el);

    if (l == 0) {
      cvt_bf16<<<(unsigned)((ns_el / 8 + 255) / 256), 256, 0, stream>>>(
          ws + OFF_XS_CUR, bf + BOFF_XSCUR, ns_el);
      cvt_bf16<<<(unsigned)((nu_el / 8 + 255) / 256), 256, 0, stream>>>(
          ws + OFF_XU_CUR, bf + BOFF_XUCUR, nu_el);
      xs_bf = bf + BOFF_XSCUR; Ks = (int)HC;
      xu_bf = bf + BOFF_XUCUR; Ku = (int)HC;
    }
  }

  // ---- 3) Outputs: mean_s [NS,128], mean_u [NU,128], out_s [NS], out_u [NU].
  float* outp   = (float*)d_out;
  float* mean_s = outp;
  float* mean_u = mean_s + NSpot * HC;
  float* out_s  = mean_u + NUser * HC;
  float* out_u  = out_s + NSpot;
  mean2<<<(unsigned)((NSpot * HC + 255) / 256), 256, 0, stream>>>(
      ws + OFF_COLL_S0, ws + OFF_COLL_S1, mean_s, NSpot * HC);
  mean2<<<(unsigned)((NUser * HC + 255) / 256), 256, 0, stream>>>(
      ws + OFF_COLL_U0, ws + OFF_COLL_U1, mean_u, NUser * HC);
  final_linear<<<(NSi + 7) / 8, 256, 0, stream>>>(ws + OFF_XS_CUR, spot_W, spot_b, out_s, NSi);
  final_linear<<<(NUi + 7) / 8, 256, 0, stream>>>(ws + OFF_XU_CUR, user_W, user_b, out_u, NUi);
}
